// QuantInvertedResidual_38336878084152
// MI455X (gfx1250) — 8-bit matrix operand order repaired and checked by reading, compiled, not run on hardware
//
#include <hip/hip_runtime.h>
#include <hip/hip_bf16.h>


typedef int  v2i __attribute__((ext_vector_type(2)));
typedef int  v4i __attribute__((ext_vector_type(4)));
typedef int  v8i __attribute__((ext_vector_type(8)));

typedef __attribute__((address_space(1))) v4i gv4i;
typedef __attribute__((address_space(3))) v4i lv4i;

#define B_     32
#define HW_    784
#define M_TOT  25088
#define CIN    96
#define CHID   576
#define COUT   96

#define OFF_W1Q   0u
#define OFF_W3Q   73728u
#define OFF_W2Q   129024u
#define OFF_PAR   138240u
#define OFF_H1Q   148480u
#define OFF_H2Q   14599168u

#define P_INVAX1 0
#define P_SCL1   1
#define P_INVAX2 2
#define P_SCL2   3
#define P_INVAX3 4
#define P_SCL3   5
#define P_BN1S   6
#define P_BN1B   582
#define P_BN2S   1158
#define P_BN2B   1734
#define P_BN3S   2310
#define P_BN3B   2406
#define P_TOTAL  2502

#if defined(__gfx1250__) && \
    __has_builtin(__builtin_amdgcn_global_load_async_to_lds_b128)
#define HAS_ASYNC_LDS 1
#else
#define HAS_ASYNC_LDS 0
#endif

__device__ __forceinline__ void wait_async_le4() {
#if defined(__gfx1250__)
#if __has_builtin(__builtin_amdgcn_s_wait_asynccnt)
  __builtin_amdgcn_s_wait_asynccnt(4);
#else
  asm volatile("s_wait_asynccnt 0x4" ::: "memory");
#endif
#endif
}
__device__ __forceinline__ void wait_async_le0() {
#if defined(__gfx1250__)
#if __has_builtin(__builtin_amdgcn_s_wait_asynccnt)
  __builtin_amdgcn_s_wait_asynccnt(0);
#else
  asm volatile("s_wait_asynccnt 0x0" ::: "memory");
#endif
#endif
}

__device__ __forceinline__ int quant4(float v, float inv_a) {
  float t = v * inv_a;
  t = fminf(fmaxf(t, -8.0f), 7.0f);
  return (int)__builtin_rintf(t);
}

__global__ __launch_bounds__(256) void prep_kernel(
    const float* __restrict__ w1, const float* __restrict__ w2,
    const float* __restrict__ w3,
    const float* __restrict__ aw1, const float* __restrict__ ax1,
    const float* __restrict__ g1, const float* __restrict__ b1,
    const float* __restrict__ m1, const float* __restrict__ v1,
    const float* __restrict__ aw2, const float* __restrict__ ax2,
    const float* __restrict__ g2, const float* __restrict__ b2,
    const float* __restrict__ m2, const float* __restrict__ v2,
    const float* __restrict__ aw3, const float* __restrict__ ax3,
    const float* __restrict__ g3, const float* __restrict__ b3,
    const float* __restrict__ m3, const float* __restrict__ v3,
    signed char* __restrict__ w1q, signed char* __restrict__ w2q,
    signed char* __restrict__ w3q, float* __restrict__ par)
{
  const int idx = blockIdx.x * 256 + threadIdx.x;
  const int S0 = 576 * 128;
  const int S1 = S0 + 96 * 576;
  const int S2 = S1 + 576 * 16;
  const int S3 = S2 + P_TOTAL;

  if (idx < S0) {
    int n = idx >> 7, k = idx & 127;
    float inv = 1.0f / aw1[0];
    w1q[idx] = (k < CIN) ? (signed char)quant4(w1[n * CIN + k], inv)
                         : (signed char)0;
  } else if (idx < S1) {
    int j = idx - S0;
    float inv = 1.0f / aw3[0];
    w3q[j] = (signed char)quant4(w3[j], inv);
  } else if (idx < S2) {
    int j = idx - S1;
    int c = j >> 4, tap = j & 15;
    float inv = 1.0f / aw2[0];
    w2q[j] = (tap < 9) ? (signed char)quant4(w2[c * 9 + tap], inv)
                       : (signed char)0;
  } else if (idx < S3) {
    int j = idx - S2;
    if (j == P_INVAX1)      par[j] = 1.0f / ax1[0];
    else if (j == P_SCL1)   par[j] = ax1[0] * aw1[0];
    else if (j == P_INVAX2) par[j] = 1.0f / ax2[0];
    else if (j == P_SCL2)   par[j] = ax2[0] * aw2[0];
    else if (j == P_INVAX3) par[j] = 1.0f / ax3[0];
    else if (j == P_SCL3)   par[j] = ax3[0] * aw3[0];
    else if (j < P_BN1B) { int c = j - P_BN1S;
      par[j] = g1[c] * __frsqrt_rn(v1[c] + 1e-5f); }
    else if (j < P_BN2S) { int c = j - P_BN1B;
      par[j] = b1[c] - m1[c] * g1[c] * __frsqrt_rn(v1[c] + 1e-5f); }
    else if (j < P_BN2B) { int c = j - P_BN2S;
      par[j] = g2[c] * __frsqrt_rn(v2[c] + 1e-5f); }
    else if (j < P_BN3S) { int c = j - P_BN2B;
      par[j] = b2[c] - m2[c] * g2[c] * __frsqrt_rn(v2[c] + 1e-5f); }
    else if (j < P_BN3B) { int c = j - P_BN3S;
      par[j] = g3[c] * __frsqrt_rn(v3[c] + 1e-5f); }
    else { int c = j - P_BN3B;
      par[j] = b3[c] - m3[c] * g3[c] * __frsqrt_rn(v3[c] + 1e-5f); }
  }
}

__global__ __launch_bounds__(128) void gemm1_kernel(
    const float* __restrict__ x, const signed char* __restrict__ w1q,
    const float* __restrict__ par, signed char* __restrict__ h1q)
{
  __shared__ __align__(16) signed char As[16 * 128];
#if HAS_ASYNC_LDS
  __shared__ __align__(16) signed char Bs[4][2][16 * 128];
#endif

  const int t = threadIdx.x;
  const int mtile = blockIdx.x;
  const float inv_ax1 = par[P_INVAX1];
  const float scale1  = par[P_SCL1];
  const float inv_ax2 = par[P_INVAX2];

  {
    const int r  = t >> 3;
    const int kg = t & 7;
    const int p  = mtile * 16 + r;
    const int b  = p / HW_, hw = p % HW_;
    const float* xp = x + ((size_t)b * CIN) * HW_ + hw;
#pragma unroll
    for (int j = 0; j < 12; ++j) {
      const int k = kg * 12 + j;
      As[r * 128 + k] = (signed char)quant4(xp[(size_t)k * HW_], inv_ax1);
    }
    *(int*)&As[r * 128 + 96 + kg * 4] = 0;
  }

  const int wave = t >> 5;
  const int lane = t & 31;
  const int hk   = lane >> 4;
  const int ln   = lane & 15;

#if HAS_ASYNC_LDS
  {
    const signed char* gsrc = w1q + (size_t)(wave * 9 * 16 + ln) * 128
                                  + hk * 64;
    signed char* ldst = &Bs[wave][0][ln * 128 + hk * 64];
#pragma unroll
    for (int c = 0; c < 4; ++c)
      __builtin_amdgcn_global_load_async_to_lds_b128(
          (gv4i*)(gsrc + c * 16), (lv4i*)(ldst + c * 16), 0, 0);
  }
#endif
  __syncthreads();

  v8i a0, a1;
  {
    const signed char* arow = &As[ln * 128];
#pragma unroll
    for (int c = 0; c < 4; ++c) {
      v2i d0 = *(const v2i*)(arow +      (c >> 1) * 32 + hk * 16 + (c & 1) * 8);
      v2i d1 = *(const v2i*)(arow + 64 + (c >> 1) * 32 + hk * 16 + (c & 1) * 8);
      a0[2 * c] = d0[0]; a0[2 * c + 1] = d0[1];
      a1[2 * c] = d1[0]; a1[2 * c + 1] = d1[1];
    }
  }

  for (int i = 0; i < 9; ++i) {
    const int ntile = wave * 9 + i;
    const int nbase = ntile * 16;

#if HAS_ASYNC_LDS
    if (i < 8) {
      const signed char* gsrc = w1q + (size_t)((ntile + 1) * 16 + ln) * 128
                                    + hk * 64;
      signed char* ldst = &Bs[wave][(i + 1) & 1][ln * 128 + hk * 64];
#pragma unroll
      for (int c = 0; c < 4; ++c)
        __builtin_amdgcn_global_load_async_to_lds_b128(
            (gv4i*)(gsrc + c * 16), (lv4i*)(ldst + c * 16), 0, 0);
      wait_async_le4();
    } else {
      wait_async_le0();
    }
    const signed char* brow = &Bs[wave][i & 1][ln * 128];
#else
    const signed char* brow = w1q + (size_t)(nbase + ln) * 128;
    if (i < 8)
      __builtin_prefetch(brow + 16 * 128, 0, 3);
#endif

    v4i q0 = *(const v4i*)(brow +       hk * 16);
    v4i q1 = *(const v4i*)(brow +  32 + hk * 16);
    v4i q2 = *(const v4i*)(brow +  64 + hk * 16);
    v4i q3 = *(const v4i*)(brow +  96 + hk * 16);
    v8i b0 = __builtin_shufflevector(q0, q1, 0, 1, 2, 3, 4, 5, 6, 7);
    v8i b1 = __builtin_shufflevector(q2, q3, 0, 1, 2, 3, 4, 5, 6, 7);

    v8i z = {0, 0, 0, 0, 0, 0, 0, 0};
    v8i acc0 = __builtin_amdgcn_wmma_i32_16x16x64_iu8(true, a0, true, b0, z,
                                                      false, false);
    v8i acc1 = __builtin_amdgcn_wmma_i32_16x16x64_iu8(true, a1, true, b1, z,
                                                      false, false);

    const int c = nbase + ln;
    const float bs = par[P_BN1S + c], bb = par[P_BN1B + c];
#pragma unroll
    for (int j = 0; j < 8; ++j) {
      const int p = mtile * 16 + j + 8 * hk;
      float f = (float)(acc0[j] + acc1[j]) * scale1;
      f = f * bs + bb;
      f = fminf(fmaxf(f, 0.0f), 6.0f);
      h1q[(size_t)p * CHID + c] = (signed char)quant4(f, inv_ax2);
    }
  }
}

__global__ __launch_bounds__(256) void dw_kernel(
    const signed char* __restrict__ h1q, const signed char* __restrict__ w2q,
    const float* __restrict__ par, signed char* __restrict__ h2q)
{
  const int idx = blockIdx.x * 256 + threadIdx.x;
  const int cg = idx % 144;
  const int p  = idx / 144;
  const int c0 = cg * 4;
  const int b  = p / HW_, hw = p % HW_;
  const int y  = hw / 28, xx = hw % 28;

  const float scale2  = par[P_SCL2];
  const float inv_ax3 = par[P_INVAX3];

  v4i wv[4];
#pragma unroll
  for (int q = 0; q < 4; ++q)
    wv[q] = *(const v4i*)(w2q + (size_t)(c0 + q) * 16);

  int acc[4] = {0, 0, 0, 0};
#pragma unroll
  for (int dy = -1; dy <= 1; ++dy) {
    const int yy = y + dy;
    if (yy < 0 || yy >= 28) continue;
#pragma unroll
    for (int dx = -1; dx <= 1; ++dx) {
      const int xc = xx + dx;
      if (xc < 0 || xc >= 28) continue;
      const unsigned v = *(const unsigned*)(
          h1q + ((size_t)b * HW_ + yy * 28 + xc) * CHID + c0);
      const int tap = (dy + 1) * 3 + (dx + 1);
#pragma unroll
      for (int q = 0; q < 4; ++q) {
        const int a = (int)(signed char)(v >> (8 * q));
        const int w = (int)(signed char)(((unsigned)wv[q][tap >> 2])
                                         >> ((tap & 3) * 8));
        acc[q] += a * w;
      }
    }
  }

  unsigned outp = 0;
#pragma unroll
  for (int q = 0; q < 4; ++q) {
    const int c = c0 + q;
    float f = (float)acc[q] * scale2;
    f = f * par[P_BN2S + c] + par[P_BN2B + c];
    f = fminf(fmaxf(f, 0.0f), 6.0f);
    outp |= ((unsigned)(quant4(f, inv_ax3) & 0xff)) << (8 * q);
  }
  *(unsigned*)(h2q + (size_t)p * CHID + c0) = outp;
}

__global__ __launch_bounds__(192) void gemm2_kernel(
    const signed char* __restrict__ h2q, const signed char* __restrict__ w3q,
    const float* __restrict__ par, const float* __restrict__ x,
    float* __restrict__ out)
{
  const int t = threadIdx.x;
  const int mtile = blockIdx.x;
  const int wave = t >> 5;
  const int lane = t & 31;
  const int hk   = lane >> 4;
  const int ln   = lane & 15;

  const signed char* arow = h2q + (size_t)(mtile * 16 + ln) * CHID;
  const signed char* brow = w3q + (size_t)(wave * 16 + ln) * CHID;
  __builtin_prefetch(arow, 0, 3);

  v8i acc0 = {0, 0, 0, 0, 0, 0, 0, 0};
  v8i acc1 = {0, 0, 0, 0, 0, 0, 0, 0};
#pragma unroll
  for (int kk = 0; kk < 9; ++kk) {
    const int kb = kk * 64;
    v8i a;
#pragma unroll
    for (int c = 0; c < 4; ++c) {
      v2i d = *(const v2i*)(arow + kb + (c >> 1) * 32 + hk * 16 + (c & 1) * 8);
      a[2 * c] = d[0]; a[2 * c + 1] = d[1];
    }
    v4i q0 = *(const v4i*)(brow + kb +      hk * 16);
    v4i q1 = *(const v4i*)(brow + kb + 32 + hk * 16);
    v8i bfrag = __builtin_shufflevector(q0, q1, 0, 1, 2, 3, 4, 5, 6, 7);
    if (kk & 1)
      acc1 = __builtin_amdgcn_wmma_i32_16x16x64_iu8(true, a, true, bfrag,
                                                    acc1, false, false);
    else
      acc0 = __builtin_amdgcn_wmma_i32_16x16x64_iu8(true, a, true, bfrag,
                                                    acc0, false, false);
  }

  const float scale3 = par[P_SCL3];
  const int c = wave * 16 + ln;
  const float bs = par[P_BN3S + c], bb = par[P_BN3B + c];
#pragma unroll
  for (int j = 0; j < 8; ++j) {
    const int p = mtile * 16 + j + 8 * hk;
    const int b = p / HW_, hw = p % HW_;
    const size_t oi = ((size_t)b * COUT + c) * HW_ + hw;
    float f = (float)(acc0[j] + acc1[j]) * scale3;
    f = f * bs + bb;
    out[oi] = f + x[oi];
  }
}

extern "C" void kernel_launch(void* const* d_in, const int* in_sizes, int n_in,
                              void* d_out, int out_size, void* d_ws,
                              size_t ws_size, hipStream_t stream) {
  const float* x   = (const float*)d_in[0];
  const float* w1  = (const float*)d_in[1];
  const float* w2  = (const float*)d_in[2];
  const float* w3  = (const float*)d_in[3];
  const float* aw1 = (const float*)d_in[4];
  const float* ax1 = (const float*)d_in[5];
  const float* g1  = (const float*)d_in[6];
  const float* b1  = (const float*)d_in[7];
  const float* m1  = (const float*)d_in[8];
  const float* v1  = (const float*)d_in[9];
  const float* aw2 = (const float*)d_in[10];
  const float* ax2 = (const float*)d_in[11];
  const float* g2  = (const float*)d_in[12];
  const float* b2  = (const float*)d_in[13];
  const float* m2  = (const float*)d_in[14];
  const float* v2  = (const float*)d_in[15];
  const float* aw3 = (const float*)d_in[16];
  const float* ax3 = (const float*)d_in[17];
  const float* g3  = (const float*)d_in[18];
  const float* b3  = (const float*)d_in[19];
  const float* m3  = (const float*)d_in[20];
  const float* v3  = (const float*)d_in[21];

  char* ws = (char*)d_ws;
  signed char* w1q = (signed char*)(ws + OFF_W1Q);
  signed char* w3q = (signed char*)(ws + OFF_W3Q);
  signed char* w2q = (signed char*)(ws + OFF_W2Q);
  float*       par = (float*)(ws + OFF_PAR);
  signed char* h1q = (signed char*)(ws + OFF_H1Q);
  signed char* h2q = (signed char*)(ws + OFF_H2Q);

  prep_kernel<<<550, 256, 0, stream>>>(
      w1, w2, w3, aw1, ax1, g1, b1, m1, v1, aw2, ax2, g2, b2, m2, v2,
      aw3, ax3, g3, b3, m3, v3, w1q, w2q, w3q, par);

  gemm1_kernel<<<M_TOT / 16, 128, 0, stream>>>(x, w1q, par, h1q);

  dw_kernel<<<(M_TOT * 144) / 256, 256, 0, stream>>>(h1q, w2q, par, h2q);

  gemm2_kernel<<<M_TOT / 16, 192, 0, stream>>>(h2q, w3q, par, x,
                                               (float*)d_out);
}
